// VectorQuantizer_43026982371847
// MI455X (gfx1250) — matrix operand order repaired and checked by reading, not compiled, not run on hardware
//
#include <hip/hip_runtime.h>

typedef __attribute__((ext_vector_type(16))) _Float16 v16h;
typedef __attribute__((ext_vector_type(8)))  float    v8f;

#define NE     1024          // codebook entries
#define CDIM   64            // channels / K
#define HW     4096          // H*W
#define NROWS  65536         // B*H*W
#define LDA    72            // padded LDS row stride in halves (144B = 9x16B -> b128-aligned)
#define OUT_ELEMS (16*64*64*64)   // 4194304

// ---------------------------------------------------------------------------
// Kernel 1: codebook fp32 -> f16, plus per-code squared norms
// ---------------------------------------------------------------------------
__global__ __launch_bounds__(256) void vq_prep(const float* __restrict__ emb,
                                               _Float16* __restrict__ emb_h,
                                               float* __restrict__ e2) {
  int j = blockIdx.x * blockDim.x + threadIdx.x;
  if (j >= NE) return;
  const float* row = emb + j * CDIM;
  float s = 0.f;
#pragma unroll
  for (int k = 0; k < CDIM; ++k) {
    float v = row[k];
    s += v * v;
    emb_h[j * CDIM + k] = (_Float16)v;
  }
  e2[j] = s;
}

// ---------------------------------------------------------------------------
// Kernel 2: WMMA distance GEMM + argmin. One block = 128 rows, 8 waves x 16.
// dist(m,n) = |e_n|^2 - 2 * z_m . e_n   (|z_m|^2 is constant per row -> dropped)
// ---------------------------------------------------------------------------
__global__ __launch_bounds__(256) void vq_argmin(const float* __restrict__ z,
                                                 const _Float16* __restrict__ emb_h,
                                                 const float* __restrict__ e2,
                                                 int* __restrict__ idx_out) {
  __shared__ _Float16 Blds[NE * LDA];     // 144 KB: full f16 codebook, 16B-aligned rows
  __shared__ _Float16 Alds[128 * LDA];    // 18 KB: transposed z tile
  __shared__ float    e2lds[NE];          // 4 KB

  const int t  = threadIdx.x;
  const int r0 = blockIdx.x * 128;

  // ---- stage codebook into LDS with 16B-granular copies (rows are 144B)
  {
    const uint4* src = (const uint4*)emb_h;   // 8192 x 16B (8 per code row)
    uint4* dst = (uint4*)Blds;                // 9 x 16B per padded row
    for (int i = t; i < NE * 8; i += 256) {
      int j  = i >> 3;          // code row
      int ch = i & 7;           // 16B chunk within row
      dst[j * 9 + ch] = src[i];
    }
  }
  for (int i = t; i < NE; i += 256) e2lds[i] = e2[i];

  // ---- stage + transpose 128 z rows: flat[r][c] = z[b][c][hw], r=(b,hw)
  const int bimg = r0 >> 12;           // 128 | HW so one image per block
  const int hw0  = r0 & (HW - 1);
  const float* zb = z + (long long)bimg * CDIM * HW + hw0;
#pragma unroll
  for (int i = 0; i < 32; ++i) {
    int linear = i * 256 + t;          // 0..8191 ; consecutive t -> consecutive hw (coalesced)
    int c    = linear >> 7;
    int rloc = linear & 127;
    Alds[rloc * LDA + c] = (_Float16)zb[c * HW + rloc];
  }
  __syncthreads();

  // ---- per-wave WMMA sweep
  const int lane  = t & 31;
  const int l15   = lane & 15;
  const int kh    = lane >> 4;         // K-half select per ISA layout
  const int mrow0 = (t >> 5) * 16;     // this wave's 16 rows (local)

  // A fragments (16x32 f16, ISA 7.12.2 layout), two K-steps covering K=64.
  // 16B-aligned 8-half chunks -> ds_load_b128.
  v16h a0, a1;
  {
    const _Float16* arow = &Alds[(mrow0 + l15) * LDA];
#pragma unroll
    for (int i = 0; i < 8; ++i) {
      a0[i]     = arow[kh * 8 + i];          // K = kh*8 + i
      a0[i + 8] = arow[16 + kh * 8 + i];     // K = 16 + kh*8 + i
      a1[i]     = arow[32 + kh * 8 + i];
      a1[i + 8] = arow[48 + kh * 8 + i];
    }
  }

  float best[8];
  int   bestn[8];
#pragma unroll
  for (int s = 0; s < 8; ++s) { best[s] = 3.4e38f; bestn[s] = 0; }

#pragma unroll 2
  for (int tile = 0; tile < NE / 16; ++tile) {
    const int n = tile * 16 + l15;                 // this lane's code column
    const _Float16* brow = &Blds[n * LDA];         // B(k,n) = emb[n][k]
    v16h b0, b1;
#pragma unroll
    for (int i = 0; i < 16; ++i) {                 // two 16B chunks, 16B aligned -> 2x ds_load_b128
      b0[i] = brow[(i >> 3) * 16 + kh * 8 + (i & 7)];       // K-step 0: K = (i>>3)*16 + kh*8 + (i&7)
      b1[i] = brow[32 + (i >> 3) * 16 + kh * 8 + (i & 7)];  // K-step 1
    }
    v8f acc = {0.f, 0.f, 0.f, 0.f, 0.f, 0.f, 0.f, 0.f};
    acc = __builtin_amdgcn_wmma_f32_16x16x32_f16(false, a0, false, b0,
                                                 (short)0, acc, false, false);
    acc = __builtin_amdgcn_wmma_f32_16x16x32_f16(false, a1, false, b1,
                                                 (short)0, acc, false, false);
    const float e2n = e2lds[n];
#pragma unroll
    for (int s = 0; s < 8; ++s) {                  // slot s -> M = s + 8*kh
      float d = __builtin_fmaf(-2.f, acc[s], e2n);
      if (d < best[s]) { best[s] = d; bestn[s] = n; }
    }
  }

  // ---- reduce argmin across the 16 lanes holding the same rows (xor 8,4,2,1)
#pragma unroll
  for (int s = 0; s < 8; ++s) {
    float v = best[s];
    int   n = bestn[s];
#pragma unroll
    for (int off = 8; off >= 1; off >>= 1) {
      float ov = __shfl_xor(v, off, 32);
      int   on = __shfl_xor(n, off, 32);
      if (ov < v || (ov == v && on < n)) { v = ov; n = on; }
    }
    best[s] = v; bestn[s] = n;
  }

  if (l15 == 0) {                                  // lanes 0 (M 0-7) and 16 (M 8-15)
    int mbase = kh * 8;
#pragma unroll
    for (int s = 0; s < 8; ++s)
      idx_out[r0 + mrow0 + mbase + s] = bestn[s];
  }
}

// ---------------------------------------------------------------------------
// Kernel 3a: zero the loss accumulator
// ---------------------------------------------------------------------------
__global__ void vq_zero_loss(float* loss) { *loss = 0.f; }

// ---------------------------------------------------------------------------
// Kernel 3b: gather z_q into [B,C,H,W] (coalesced writes) + fused loss reduce
// ---------------------------------------------------------------------------
__global__ __launch_bounds__(256) void vq_scatter(const float* __restrict__ z,
                                                  const float* __restrict__ emb,
                                                  const int* __restrict__ idx,
                                                  float* __restrict__ out,
                                                  float* __restrict__ loss) {
  __shared__ float red[256];
  const int tid = threadIdx.x;
  const int t   = blockIdx.x * 256 + tid;          // linear index into [B,C,H,W]
  const int hw  = t & (HW - 1);
  const int c   = (t >> 12) & (CDIM - 1);
  const int b   = t >> 18;
  const int r   = (b << 12) | hw;                  // flat row id

  float q = emb[idx[r] * CDIM + c];                // codebook is L2-resident
  float d = q - z[t];
  out[t] = q;

  red[tid] = d * d;
  __syncthreads();
#pragma unroll
  for (int s = 128; s > 0; s >>= 1) {
    if (tid < s) red[tid] += red[tid + s];
    __syncthreads();
  }
  if (tid == 0)
    atomicAdd(loss, red[0] * (11.f / (float)OUT_ELEMS));  // (1+KL_SCALE)*MSE
}

// ---------------------------------------------------------------------------
extern "C" void kernel_launch(void* const* d_in, const int* in_sizes, int n_in,
                              void* d_out, int out_size, void* d_ws, size_t ws_size,
                              hipStream_t stream) {
  const float* z   = (const float*)d_in[0];        // [16,64,64,64] fp32
  const float* emb = (const float*)d_in[1];        // [1024,64] fp32

  float* out  = (float*)d_out;                     // 4194304 floats
  float* loss = out + OUT_ELEMS;                   // 1 float
  int*   idx  = (int*)(out + OUT_ELEMS + 1);       // 65536 int32 (bit-exact)

  _Float16* emb_h = (_Float16*)d_ws;                       // 128 KB
  float*    e2    = (float*)((char*)d_ws + NE * CDIM * sizeof(_Float16));

  vq_prep   <<<NE / 256, 256, 0, stream>>>(emb, emb_h, e2);
  vq_argmin <<<NROWS / 128, 256, 0, stream>>>(z, emb_h, e2, idx);
  vq_zero_loss<<<1, 1, 0, stream>>>(loss);
  vq_scatter<<<OUT_ELEMS / 256, 256, 0, stream>>>(z, emb, idx, out, loss);
}
